// EdgePredictor_59167469470093
// MI455X (gfx1250) — matrix operand order repaired and checked by reading, not compiled, not run on hardware
//
#include <hip/hip_runtime.h>

typedef __attribute__((ext_vector_type(16))) _Float16 v16h;
typedef __attribute__((ext_vector_type(8)))  _Float16 v8h;
typedef __attribute__((ext_vector_type(8)))  float    v8f;
typedef __attribute__((ext_vector_type(4)))  float    v4f;

#define DIM 128          // node/edge feature dim
#define K1  384          // 3*DIM, layer-1 K
#define HID 64           // hidden dim
#define WAVES_PER_BLOCK 4
#define BLOCK_T (WAVES_PER_BLOCK * 32)
#define NFRAG1 48        // 12 K-steps * 4 N-tiles for W1
#define FRAG_H 16        // f16 elements per lane per fragment

__global__ __launch_bounds__(BLOCK_T)
void edge_mlp_wmma(const int* __restrict__ eidx, const float* __restrict__ x,
                   const float* __restrict__ e,  const float* __restrict__ W1,
                   const float* __restrict__ b1, const float* __restrict__ W2,
                   const float* __restrict__ b2, const float* __restrict__ W3,
                   const float* __restrict__ b3, float* __restrict__ out,
                   int E, int tiles)
{
    // 48KB: W1 packed as WMMA B fragments (f16), shared by all waves in block
    __shared__ _Float16 lds_w1[NFRAG1 * 32 * FRAG_H];
    // 8KB: per-wave 16x64 f16 staging tile for layer1->layer2 transpose
    __shared__ _Float16 lds_h[WAVES_PER_BLOCK * 16 * HID];

    const int tid  = threadIdx.x;
    const int lane = tid & 31;
    const int wave = tid >> 5;
    const int lo   = lane & 15;
    const int hi   = lane >> 4;

    // ---- one-time: pack W1 into LDS in B-fragment layout ----
    // frag f = kk*4 + nt ; element (laneSlot, j):
    //   n = nt*16 + (laneSlot&15), k = kk*32 + (laneSlot>>4)*16 + j
    for (int idx = tid; idx < NFRAG1 * 32 * FRAG_H; idx += BLOCK_T) {
        int f   = idx >> 9;        // / 512
        int rem = idx & 511;
        int fl  = rem >> 4;
        int j   = rem & 15;
        int n   = ((f & 3) << 4) + (fl & 15);
        int k   = ((f >> 2) << 5) + ((fl >> 4) << 4) + j;
        lds_w1[idx] = (_Float16)W1[n * K1 + k];
    }
    __syncthreads();

    // ---- per-wave: W2 B-fragments live in registers (contiguous 16-float runs) ----
    v16h w2f[8];
#pragma unroll
    for (int f = 0; f < 8; ++f) {
        int n  = ((f & 3) << 4) + lo;
        int kb = ((f >> 2) << 5) + (hi << 4);
        const v4f* p = (const v4f*)(W2 + n * HID + kb);
        v4f q0 = p[0], q1 = p[1], q2 = p[2], q3 = p[3];
        v16h t;
#pragma unroll
        for (int j = 0; j < 4; ++j) {
            t[j]      = (_Float16)q0[j];
            t[4 + j]  = (_Float16)q1[j];
            t[8 + j]  = (_Float16)q2[j];
            t[12 + j] = (_Float16)q3[j];
        }
        w2f[f] = t;
    }

    // per-lane constants (C layout: col = nt*16 + lo)
    const float b1v[4] = { b1[lo], b1[16 + lo], b1[32 + lo], b1[48 + lo] };
    const float b2v[4] = { b2[lo], b2[16 + lo], b2[32 + lo], b2[48 + lo] };
    const float w3v[4] = { W3[lo], W3[16 + lo], W3[32 + lo], W3[48 + lo] };
    const float b3v = b3[0];

    _Float16* hw = lds_h + wave * (16 * HID);

    const int gwave   = blockIdx.x * WAVES_PER_BLOCK + wave;
    const int wstride = gridDim.x * WAVES_PER_BLOCK;

    for (int tile = gwave; tile < tiles; tile += wstride) {
        const int ebase = tile << 4;
        int myedge = ebase + lo;
        if (myedge >= E) myedge = E - 1;           // clamp (keeps EXEC full for WMMA)
        const int s = eidx[myedge];
        const int d = eidx[E + myedge];
        const float* px = x + (long)s * DIM;
        const float* pd = x + (long)d * DIM;
        const float* pe = e + (long)myedge * DIM;

        // prefetch next tile's streaming edge-feature row
        {
            long nxt = (long)(tile + wstride) * 16 + lo;
            if (nxt < (long)E) __builtin_prefetch(e + nxt * DIM, 0, 0);
        }

        // ---- layer 1: [16 x 384] x [384 x 64], 12 K-steps x 4 N-tiles ----
        v8f acc[4];
#pragma unroll
        for (int nt = 0; nt < 4; ++nt)
#pragma unroll
            for (int j = 0; j < 8; ++j) acc[nt][j] = b1v[nt];

#pragma unroll
        for (int kk = 0; kk < 12; ++kk) {
            // A fragment, contiguous k order as in B: lane<16 -> K = base+0..15
            //                               lane>=16 -> K = base+16..31
            int c0 = (kk << 5) + (hi << 4);
            int c1 = c0 + 8;
            const float* q0 = (c0 < DIM) ? (px + c0)
                             : (c0 < 2 * DIM) ? (pd + c0 - DIM) : (pe + c0 - 2 * DIM);
            const float* q1 = (c1 < DIM) ? (px + c1)
                             : (c1 < 2 * DIM) ? (pd + c1 - DIM) : (pe + c1 - 2 * DIM);
            v4f f0 = ((const v4f*)q0)[0], f1 = ((const v4f*)q0)[1];
            v4f f2 = ((const v4f*)q1)[0], f3 = ((const v4f*)q1)[1];
            v16h a;
#pragma unroll
            for (int j = 0; j < 4; ++j) {
                a[j]      = (_Float16)f0[j];
                a[4 + j]  = (_Float16)f1[j];
                a[8 + j]  = (_Float16)f2[j];
                a[12 + j] = (_Float16)f3[j];
            }
            const _Float16* wbase = lds_w1 + ((size_t)(kk * 4) * 32 + lane) * FRAG_H;
#pragma unroll
            for (int nt = 0; nt < 4; ++nt) {
                v16h b = *(const v16h*)(wbase + (size_t)nt * 32 * FRAG_H);
                acc[nt] = __builtin_amdgcn_wmma_f32_16x16x32_f16(
                    false, a, false, b, (short)0, acc[nt], false, false);
            }
        }

        // ---- relu + stage h1 to LDS (C layout -> row-major f16 tile) ----
#pragma unroll
        for (int nt = 0; nt < 4; ++nt)
#pragma unroll
            for (int r = 0; r < 8; ++r) {
                float v = acc[nt][r];
                v = v > 0.f ? v : 0.f;
                hw[(r + (hi << 3)) * HID + (nt << 4) + lo] = (_Float16)v;
            }
        asm volatile("s_wait_dscnt 0x0" ::: "memory");

        // ---- layer 2: [16 x 64] x [64 x 64], 2 K-steps x 4 N-tiles ----
        v8f acc2[4];
#pragma unroll
        for (int nt = 0; nt < 4; ++nt)
#pragma unroll
            for (int j = 0; j < 8; ++j) acc2[nt][j] = b2v[nt];

#pragma unroll
        for (int kk = 0; kk < 2; ++kk) {
            const _Float16* s0 = hw + lo * HID + (kk << 5) + (hi << 4);
            v8h l8 = *(const v8h*)s0;
            v8h h8 = *(const v8h*)(s0 + 8);
            v16h a;
#pragma unroll
            for (int j = 0; j < 8; ++j) { a[j] = l8[j]; a[8 + j] = h8[j]; }
#pragma unroll
            for (int nt = 0; nt < 4; ++nt) {
                acc2[nt] = __builtin_amdgcn_wmma_f32_16x16x32_f16(
                    false, a, false, w2f[kk * 4 + nt], (short)0, acc2[nt], false, false);
            }
        }

        // ---- layer 3: relu(h2) . W3 + b3, VALU + cross-lane reduce ----
        float part[8];
#pragma unroll
        for (int r = 0; r < 8; ++r) {
            float v0 = fmaxf(acc2[0][r], 0.f);
            float v1 = fmaxf(acc2[1][r], 0.f);
            float v2 = fmaxf(acc2[2][r], 0.f);
            float v3 = fmaxf(acc2[3][r], 0.f);
            part[r] = v0 * w3v[0] + v1 * w3v[1] + v2 * w3v[2] + v3 * w3v[3];
        }
#pragma unroll
        for (int r = 0; r < 8; ++r) {
            float p = part[r];
            p += __shfl_xor(p, 1, 32);
            p += __shfl_xor(p, 2, 32);
            p += __shfl_xor(p, 4, 32);
            p += __shfl_xor(p, 8, 32);
            part[r] = p;
        }
        if (lo == 0) {  // lanes 0 (rows 0-7) and 16 (rows 8-15)
#pragma unroll
            for (int r = 0; r < 8; ++r) {
                int row = ebase + (hi << 3) + r;
                if (row < E) out[row] = part[r] + b3v;
            }
        }
    }
}

extern "C" void kernel_launch(void* const* d_in, const int* in_sizes, int n_in,
                              void* d_out, int out_size, void* d_ws, size_t ws_size,
                              hipStream_t stream) {
    const int*   eidx = (const int*)  d_in[0];
    const float* x    = (const float*)d_in[1];
    const float* e    = (const float*)d_in[2];
    const float* W1   = (const float*)d_in[3];
    const float* b1   = (const float*)d_in[4];
    const float* W2   = (const float*)d_in[5];
    const float* b2   = (const float*)d_in[6];
    const float* W3   = (const float*)d_in[7];
    const float* b3   = (const float*)d_in[8];
    float* out = (float*)d_out;

    const int E     = in_sizes[0] / 2;       // edge_index is [2, E]
    const int tiles = (E + 15) >> 4;
    if (tiles <= 0) return;

    int grid = (tiles + WAVES_PER_BLOCK - 1) / WAVES_PER_BLOCK;
    if (grid > 2048) grid = 2048;

    edge_mlp_wmma<<<grid, BLOCK_T, 0, stream>>>(eidx, x, e, W1, b1, W2, b2, W3, b3,
                                                out, E, tiles);
}
